// RecurrentNeuralNetwork_72576357368454
// MI455X (gfx1250) — matrix operand order repaired and checked by reading, not compiled, not run on hardware
//
#include <hip/hip_runtime.h>
#include <hip/hip_bf16.h>

// ---------------------------------------------------------------------------
// Types for CDNA5 WMMA (wave32): v_wmma_f32_16x16x32_bf16
// ---------------------------------------------------------------------------
typedef __attribute__((ext_vector_type(16))) __bf16 v16bf;
typedef __attribute__((ext_vector_type(8)))  float  v8f;
typedef __attribute__((ext_vector_type(4)))  unsigned int v4u;
typedef __attribute__((ext_vector_type(8)))  unsigned int v8u;

union ABFrag {
    v16bf v;
    uint4 q[2];
    unsigned short s[16];
};

__device__ __forceinline__ unsigned short f32_to_bf16(float f) {
    unsigned int u = __float_as_uint(f);
    unsigned int r = u + 0x7FFFu + ((u >> 16) & 1u);   // round-to-nearest-even
    return (unsigned short)(r >> 16);
}
__device__ __forceinline__ float bf16u_to_f32(unsigned short h) {
    return __uint_as_float(((unsigned int)h) << 16);
}
__device__ __forceinline__ float sigmoidf_(float x) {
    return 1.0f / (1.0f + __expf(-x));
}
__device__ __forceinline__ float tanhf_(float x) {
    float e = __expf(-2.0f * x);
    return (1.0f - e) / (1.0f + e);
}

// ---------------------------------------------------------------------------
// Tensor Data Mover: 1-D tile load Global -> LDS.
//   nbytes must be a multiple of 4 and <= 65535*4.  Issue from one wave,
//   then s_wait_tensorcnt + workgroup barrier before any LDS reads.
// D# layout per CDNA5 ISA 8.3/8.4:
//   g0: [1:0]=count(1), [63:32]=lds_addr, [120:64]=global_addr, [127:126]=type(2)
//   g1: [17:16]=data_size(2 => 4B), [79:48]=tensor_dim0, [111:80]=tensor_dim1,
//       [127:112]=tile_dim0, [143:128]=tile_dim1, [207:160]=tensor_dim0_stride
// ---------------------------------------------------------------------------
__device__ __forceinline__ void tdm_load_to_lds_1d(const void* gptr,
                                                   unsigned lds_byte_off,
                                                   unsigned nbytes) {
    const unsigned long long ga = (unsigned long long)(size_t)gptr;
    const unsigned n4 = nbytes >> 2;                    // 4-byte elements

    v4u g0;
    g0[0] = 1u;                                         // count=1, user descriptor
    g0[1] = lds_byte_off;                               // lds_addr
    g0[2] = (unsigned)(ga & 0xFFFFFFFFu);               // global_addr[31:0]
    g0[3] = (unsigned)((ga >> 32) & 0x01FFFFFFu)        // global_addr[56:32]
          | (2u << 30);                                 // type = 2 ("image")

    v8u g1;
    g1[0] = (2u << 16);                                 // wg_mask=0, data_size=4B
    g1[1] = (n4 & 0xFFFFu) << 16;                       // tensor_dim0[15:0]
    g1[2] = ((n4 >> 16) & 0xFFFFu) | (1u << 16);        // tensor_dim0[31:16], tensor_dim1=1
    g1[3] = (n4 & 0xFFFFu) << 16;                       // tile_dim0 = n4
    g1[4] = 1u;                                         // tile_dim1 = 1
    g1[5] = n4;                                         // tensor_dim0_stride[31:0]
    g1[6] = 0u;
    g1[7] = 0u;

    asm volatile("tensor_load_to_lds %0, %1"
                 :: "s"(g0), "s"(g1)
                 : "memory");
}

// Problem constants
#define BB   32
#define TT   512
#define HH   1024
#define G4H  4096
#define TCH  128          // T-chunk precomputed per GEMM launch

// ---------------------------------------------------------------------------
// Elementwise f32 -> bf16 convert (weights, input activations)
// ---------------------------------------------------------------------------
__global__ void cvt_f32_to_bf16_kernel(const float* __restrict__ src,
                                       unsigned short* __restrict__ dst, int n) {
    int stride = gridDim.x * blockDim.x;
    for (int i = blockIdx.x * blockDim.x + threadIdx.x; i < n; i += stride)
        dst[i] = f32_to_bf16(src[i]);
}

// Zero recurrent state (c, h_f32, h_bf16 ping buffer)
__global__ void zero_state_kernel(float* __restrict__ c, float* __restrict__ h,
                                  unsigned short* __restrict__ hb, int n) {
    int stride = gridDim.x * blockDim.x;
    for (int i = blockIdx.x * blockDim.x + threadIdx.x; i < n; i += stride) {
        c[i] = 0.0f; h[i] = 0.0f; hb[i] = 0;
    }
}

// Copy final h / c states into d_out tail sections
__global__ void copy_final_kernel(const float* __restrict__ h, const float* __restrict__ c,
                                  float* __restrict__ out_h, float* __restrict__ out_c, int n) {
    int stride = gridDim.x * blockDim.x;
    for (int i = blockIdx.x * blockDim.x + threadIdx.x; i < n; i += stride) {
        out_h[i] = h[i]; out_c[i] = c[i];
    }
}

// ---------------------------------------------------------------------------
// Input projection GEMM:  XG[r, j] = sum_k Xbf[row r, k] * Wih[j, k] + bih[j]
//   rows r = t_local*32 + b   (t_local in [0,TCH), b in [0,32))
//   Xbf is [B, T, H] bf16; XG is [TCH, B, 4H] f32.
// Grid: (M/64, N/128), block 256 (8 waves). Each wave: 32x32 tile, K-loop of 32.
// ---------------------------------------------------------------------------
__global__ void __launch_bounds__(256)
lstm_xg_gemm_kernel(const unsigned short* __restrict__ X,
                    const unsigned short* __restrict__ W,
                    const float* __restrict__ bih,
                    float* __restrict__ XG, int t0) {
    const int lane = threadIdx.x & 31;
    const int wave = threadIdx.x >> 5;           // 0..7
    const int Mbase = blockIdx.x * 64 + (wave >> 2) * 32;   // 2 M sub-tiles
    const int Nbase = blockIdx.y * 128 + (wave & 3) * 32;   // 4 N sub-tiles

    const int hb = lane >> 4;                    // 0/1 half select
    const int lr = lane & 15;

    // Row pointers for the two 16-row A tiles (row -> (b,t) swizzle of [B,T,H])
    int r0 = Mbase + lr;
    int r1 = r0 + 16;
    const unsigned short* aptr0 = X + ((size_t)(r0 & 31) * TT + (t0 + (r0 >> 5))) * HH;
    const unsigned short* aptr1 = X + ((size_t)(r1 & 31) * TT + (t0 + (r1 >> 5))) * HH;
    // Row pointers for the two 16-col B tiles (rows of Wih)
    const unsigned short* bptr0 = W + (size_t)(Nbase + lr) * HH;
    const unsigned short* bptr1 = W + (size_t)(Nbase + 16 + lr) * HH;

    v8f c00 = {}; v8f c01 = {}; v8f c10 = {}; v8f c11 = {};

    for (int kk = 0; kk < HH / 32; ++kk) {
        const int k0 = kk * 32;
        const int ka = k0 + hb * 8;      // A: halves at ka and ka+16
        const int kb = k0 + hb * 8;      // B: halves at kb and kb+16

        ABFrag a0, a1, b0, b1;
        a0.q[0] = *(const uint4*)(aptr0 + ka);
        a0.q[1] = *(const uint4*)(aptr0 + ka + 16);
        a1.q[0] = *(const uint4*)(aptr1 + ka);
        a1.q[1] = *(const uint4*)(aptr1 + ka + 16);
        b0.q[0] = *(const uint4*)(bptr0 + kb);
        b0.q[1] = *(const uint4*)(bptr0 + kb + 16);
        b1.q[0] = *(const uint4*)(bptr1 + kb);
        b1.q[1] = *(const uint4*)(bptr1 + kb + 16);
        __builtin_prefetch(bptr0 + kb + 64, 0, 3);
        __builtin_prefetch(bptr1 + kb + 64, 0, 3);

        c00 = __builtin_amdgcn_wmma_f32_16x16x32_bf16(false, a0.v, false, b0.v, (short)0, c00, false, false);
        c01 = __builtin_amdgcn_wmma_f32_16x16x32_bf16(false, a0.v, false, b1.v, (short)0, c01, false, false);
        c10 = __builtin_amdgcn_wmma_f32_16x16x32_bf16(false, a1.v, false, b0.v, (short)0, c10, false, false);
        c11 = __builtin_amdgcn_wmma_f32_16x16x32_bf16(false, a1.v, false, b1.v, (short)0, c11, false, false);
    }

    // Epilogue: D element (VGPR v, lane) -> m = 8*hb + v, n = lane&15
    #pragma unroll
    for (int v = 0; v < 8; ++v) {
        int m0 = Mbase + hb * 8 + v;
        int m1 = m0 + 16;
        int n0 = Nbase + lr;
        int n1 = n0 + 16;
        XG[(size_t)m0 * G4H + n0] = c00[v] + bih[n0];
        XG[(size_t)m0 * G4H + n1] = c01[v] + bih[n1];
        XG[(size_t)m1 * G4H + n0] = c10[v] + bih[n0];
        XG[(size_t)m1 * G4H + n1] = c11[v] + bih[n1];
    }
}

// ---------------------------------------------------------------------------
// One LSTM timestep: gates = xg_t + h_{t-1}·Whhᵀ + bhh ; cell update.
// Block = 128 (4 waves), grid = 16.  Each wave owns 16 hidden columns and
// computes all 4 gate blocks for them (i: j, f: j+1024, g: j+2048, o: j+3072),
// so the nonlinearity is lane-local.  h_{t-1} (bf16, 64KB) staged in LDS by
// the Tensor Data Mover (one DMA per workgroup, TENSORcnt-tracked).
// ---------------------------------------------------------------------------
__global__ void __launch_bounds__(128)
lstm_step_kernel(const unsigned short* __restrict__ h_in,   // [32,1024] bf16
                 unsigned short* __restrict__ h_out,        // [32,1024] bf16
                 float* __restrict__ c_state,               // [32,1024]
                 float* __restrict__ h_state,               // [32,1024]
                 const float* __restrict__ xg_t,            // [32,4096]
                 const unsigned short* __restrict__ Whh,    // [4096,1024] bf16
                 const float* __restrict__ bhh,             // [4096]
                 unsigned short* __restrict__ sink_bf,      // layer0: x1 + t*H (stride B = T*H) or null
                 float* __restrict__ sink_f32)              // layer1: out + t*H (stride B = T*H) or null
{
    __shared__ __align__(16) unsigned short sh[BB * HH];    // 64 KB

    const int tid  = threadIdx.x;
    const int lane = tid & 31;
    const int wave = tid >> 5;                  // 0..3
    const int hb   = lane >> 4;
    const int lr   = lane & 15;
    const int n0   = blockIdx.x * 64 + wave * 16;   // hidden-column base

    // TDM stage of h_{t-1} into LDS: one DMA issued by wave 0.
    if (wave == 0) {
        tdm_load_to_lds_1d(h_in, (unsigned)(size_t)sh, BB * HH * 2);
        __builtin_amdgcn_s_wait_tensorcnt(0);
    }
    __syncthreads();

    // Weight row pointers: gate g lives at Whh row g*1024 + n
    const unsigned short* bp0 = Whh + (size_t)(0 * HH + n0 + lr) * HH;
    const unsigned short* bp1 = Whh + (size_t)(1 * HH + n0 + lr) * HH;
    const unsigned short* bp2 = Whh + (size_t)(2 * HH + n0 + lr) * HH;
    const unsigned short* bp3 = Whh + (size_t)(3 * HH + n0 + lr) * HH;

    // A tile row pointers in LDS (two 16-row tiles cover batch 0..31)
    const unsigned short* ap0 = sh + (size_t)(lr) * HH;
    const unsigned short* ap1 = sh + (size_t)(lr + 16) * HH;

    v8f acc[4][2];
    #pragma unroll
    for (int g = 0; g < 4; ++g) { acc[g][0] = (v8f){}; acc[g][1] = (v8f){}; }

    for (int kk = 0; kk < HH / 32; ++kk) {
        const int k0 = kk * 32;
        const int ka = k0 + hb * 8;
        const int kb = k0 + hb * 8;

        ABFrag a0, a1;
        a0.q[0] = *(const uint4*)(ap0 + ka);
        a0.q[1] = *(const uint4*)(ap0 + ka + 16);
        a1.q[0] = *(const uint4*)(ap1 + ka);
        a1.q[1] = *(const uint4*)(ap1 + ka + 16);

        ABFrag b0, b1, b2, b3;
        b0.q[0] = *(const uint4*)(bp0 + kb);  b0.q[1] = *(const uint4*)(bp0 + kb + 16);
        b1.q[0] = *(const uint4*)(bp1 + kb);  b1.q[1] = *(const uint4*)(bp1 + kb + 16);
        b2.q[0] = *(const uint4*)(bp2 + kb);  b2.q[1] = *(const uint4*)(bp2 + kb + 16);
        b3.q[0] = *(const uint4*)(bp3 + kb);  b3.q[1] = *(const uint4*)(bp3 + kb + 16);
        __builtin_prefetch(bp0 + kb + 64, 0, 3);
        __builtin_prefetch(bp1 + kb + 64, 0, 3);
        __builtin_prefetch(bp2 + kb + 64, 0, 3);
        __builtin_prefetch(bp3 + kb + 64, 0, 3);

        acc[0][0] = __builtin_amdgcn_wmma_f32_16x16x32_bf16(false, a0.v, false, b0.v, (short)0, acc[0][0], false, false);
        acc[0][1] = __builtin_amdgcn_wmma_f32_16x16x32_bf16(false, a1.v, false, b0.v, (short)0, acc[0][1], false, false);
        acc[1][0] = __builtin_amdgcn_wmma_f32_16x16x32_bf16(false, a0.v, false, b1.v, (short)0, acc[1][0], false, false);
        acc[1][1] = __builtin_amdgcn_wmma_f32_16x16x32_bf16(false, a1.v, false, b1.v, (short)0, acc[1][1], false, false);
        acc[2][0] = __builtin_amdgcn_wmma_f32_16x16x32_bf16(false, a0.v, false, b2.v, (short)0, acc[2][0], false, false);
        acc[2][1] = __builtin_amdgcn_wmma_f32_16x16x32_bf16(false, a1.v, false, b2.v, (short)0, acc[2][1], false, false);
        acc[3][0] = __builtin_amdgcn_wmma_f32_16x16x32_bf16(false, a0.v, false, b3.v, (short)0, acc[3][0], false, false);
        acc[3][1] = __builtin_amdgcn_wmma_f32_16x16x32_bf16(false, a1.v, false, b3.v, (short)0, acc[3][1], false, false);
    }

    // Cell update: lane-local across the 4 gates.
    const int nh = n0 + lr;
    const float bi = bhh[0 * HH + nh];
    const float bf = bhh[1 * HH + nh];
    const float bg = bhh[2 * HH + nh];
    const float bo = bhh[3 * HH + nh];

    #pragma unroll
    for (int mt = 0; mt < 2; ++mt) {
        #pragma unroll
        for (int v = 0; v < 8; ++v) {
            const int m = mt * 16 + hb * 8 + v;      // batch index 0..31
            const float* xg = xg_t + (size_t)m * G4H;
            float ig = sigmoidf_(acc[0][mt][v] + xg[0 * HH + nh] + bi);
            float fg = sigmoidf_(acc[1][mt][v] + xg[1 * HH + nh] + bf);
            float gg = tanhf_   (acc[2][mt][v] + xg[2 * HH + nh] + bg);
            float og = sigmoidf_(acc[3][mt][v] + xg[3 * HH + nh] + bo);

            const size_t idx = (size_t)m * HH + nh;
            float cn = fg * c_state[idx] + ig * gg;
            float hn = og * tanhf_(cn);
            c_state[idx] = cn;
            h_state[idx] = hn;
            h_out[idx]   = f32_to_bf16(hn);
            if (sink_bf)  sink_bf [(size_t)m * (TT * HH) + nh] = f32_to_bf16(hn);
            if (sink_f32) sink_f32[(size_t)m * (TT * HH) + nh] = hn;
        }
    }
}

// ---------------------------------------------------------------------------
// Host launcher
// ---------------------------------------------------------------------------
extern "C" void kernel_launch(void* const* d_in, const int* in_sizes, int n_in,
                              void* d_out, int out_size, void* d_ws, size_t ws_size,
                              hipStream_t stream) {
    (void)in_sizes; (void)n_in; (void)out_size; (void)ws_size;

    const float* inp = (const float*)d_in[0];   // [32,512,1024]
    const float* Wih = (const float*)d_in[1];   // [2,4096,1024]
    const float* Whh = (const float*)d_in[2];   // [2,4096,1024]
    const float* bih = (const float*)d_in[3];   // [2,4096]
    const float* bhh = (const float*)d_in[4];   // [2,4096]
    float* out = (float*)d_out;                 // prev | final_h | final_c

    // ---- workspace carve (256B aligned) ----
    char* base = (char*)d_ws;
    size_t off = 0;
    auto carve = [&](size_t bytes) -> char* {
        char* p = base + off;
        off += (bytes + 255) & ~(size_t)255;
        return p;
    };
    unsigned short* wih_bf = (unsigned short*)carve((size_t)2 * G4H * HH * 2);  // 16 MB
    unsigned short* whh_bf = (unsigned short*)carve((size_t)2 * G4H * HH * 2);  // 16 MB
    unsigned short* x0     = (unsigned short*)carve((size_t)BB * TT * HH * 2);  // 32 MB
    unsigned short* x1     = (unsigned short*)carve((size_t)BB * TT * HH * 2);  // 32 MB
    float*          xg     = (float*)        carve((size_t)TCH * BB * G4H * 4); // 64 MB
    float*          c_st   = (float*)        carve((size_t)BB * HH * 4);
    float*          h_st   = (float*)        carve((size_t)BB * HH * 4);
    unsigned short* hb0    = (unsigned short*)carve((size_t)BB * HH * 2);
    unsigned short* hb1    = (unsigned short*)carve((size_t)BB * HH * 2);

    // ---- 1) f32 -> bf16 conversions ----
    cvt_f32_to_bf16_kernel<<<2048, 256, 0, stream>>>(Wih, wih_bf, 2 * G4H * HH);
    cvt_f32_to_bf16_kernel<<<2048, 256, 0, stream>>>(Whh, whh_bf, 2 * G4H * HH);
    cvt_f32_to_bf16_kernel<<<2048, 256, 0, stream>>>(inp, x0, BB * TT * HH);

    // ---- 2) layers ----
    for (int l = 0; l < 2; ++l) {
        zero_state_kernel<<<64, 256, 0, stream>>>(c_st, h_st, hb0, BB * HH);

        const unsigned short* xs   = (l == 0) ? x0 : x1;
        const unsigned short* Wl   = wih_bf + (size_t)l * G4H * HH;
        const unsigned short* Whl  = whh_bf + (size_t)l * G4H * HH;
        const float*          bihl = bih + l * G4H;
        const float*          bhhl = bhh + l * G4H;

        int p = 0;
        for (int t0 = 0; t0 < TT; t0 += TCH) {
            // Input projection for this chunk: [TCH*B, 4H] = Xbf · Wihᵀ + bih
            dim3 ggrid((TCH * BB) / 64, G4H / 128);
            lstm_xg_gemm_kernel<<<ggrid, 256, 0, stream>>>(xs, Wl, bihl, xg, t0);

            for (int tl = 0; tl < TCH; ++tl) {
                const int t = t0 + tl;
                unsigned short* hin  = p ? hb1 : hb0;
                unsigned short* hout = p ? hb0 : hb1;
                unsigned short* sbf  = (l == 0) ? (x1 + (size_t)t * HH) : nullptr;
                float*          sf   = (l == 1) ? (out + (size_t)t * HH) : nullptr;
                lstm_step_kernel<<<16, 128, 0, stream>>>(
                    hin, hout, c_st, h_st,
                    xg + (size_t)tl * BB * G4H, Whl, bhhl, sbf, sf);
                p ^= 1;
            }
        }
        // final h / c for this layer
        copy_final_kernel<<<64, 256, 0, stream>>>(
            h_st, c_st,
            out + (size_t)BB * TT * HH + (size_t)l * BB * HH,
            out + (size_t)BB * TT * HH + (size_t)2 * BB * HH + (size_t)l * BB * HH,
            BB * HH);
    }
}
